// EncoderDecoderWrapper_23467701305756
// MI455X (gfx1250) — compile-verified, matrix operand order checked by reading, not run on hardware
//
#include <hip/hip_runtime.h>

// ---------------- types ----------------
typedef __bf16 bf16;
typedef __attribute__((ext_vector_type(8)))  __bf16 v8bf;
typedef __attribute__((ext_vector_type(16))) __bf16 v16bf;
typedef __attribute__((ext_vector_type(8)))  float  v8f;

// ---------------- dims ----------------
#define N_BATCH 2048
#define LX      168
#define FF      64
#define HIDN    256
#define G3      768      // 3 * HID
#define LY      72
#define FT      22
#define FOUTN   10
#define DEC_INW 32
#define REG_H   128

__device__ __forceinline__ unsigned short f32_to_bf16_rne(float f) {
    unsigned int u = __float_as_uint(f);
    unsigned int r = u + 0x7FFFu + ((u >> 16) & 1u);
    return (unsigned short)(r >> 16);
}

__device__ __forceinline__ float smelu_f(float x) {
    const float B = 1.1f;
    if (x >= B) return x;
    if (x <= -B) return 0.0f;
    float v = x + B;
    return v * v * (1.0f / (4.0f * B));
}

// ---------------- generic f32 -> bf16 convert ----------------
__global__ void cvt_bf16_kernel(const float* __restrict__ src,
                                unsigned short* __restrict__ dst, long n) {
    long i = (long)blockIdx.x * blockDim.x + threadIdx.x;
    if (i < n) dst[i] = f32_to_bf16_rne(src[i]);
}

// ---------------- zero h (f32 + bf16) ----------------
__global__ void zero_h_kernel(float* __restrict__ h, unsigned short* __restrict__ hb) {
    int i = blockIdx.x * blockDim.x + threadIdx.x;
    if (i < N_BATCH * HIDN) { h[i] = 0.0f; hb[i] = 0; }
}

// ---------------- pack decoder inputs: (Ly, N, 32) bf16 ----------------
// dec_in[t] = cat(y_t[:, s, :22], p[:, s, :10]) with s = (t==0 ? 0 : t-1)
__global__ void pack_dec_kernel(const float* __restrict__ y_t,
                                const float* __restrict__ p,
                                unsigned short* __restrict__ dec) {
    long idx = (long)blockIdx.x * blockDim.x + threadIdx.x;
    const long total = (long)LY * N_BATCH * DEC_INW;
    if (idx >= total) return;
    int t = (int)(idx / (N_BATCH * DEC_INW));
    int r = (int)(idx - (long)t * (N_BATCH * DEC_INW));
    int b = r >> 5;
    int f = r & 31;
    int s = (t == 0) ? 0 : (t - 1);
    float v;
    if (f < FT) v = y_t[((long)b * (LY + 1) + s) * FT + f];
    else        v = p  [((long)b * (LY + 1) + s) * FOUTN + (f - FT)];
    dec[idx] = f32_to_bf16_rne(v);
}

// ---------------- dual-descriptor WMMA GEMM ----------------
// C = A(M x K, bf16, row stride lda) @ W^T   (W is (Ncols x K) bf16 row-major)
// Workgroup = 256 threads = 8 waves. Wave (msub,nsub) computes a 16x64 tile.
// Workgroup tile: 32 rows x 256 cols. grid.x = M/32, grid.y = tiles0 + tiles1.
// blockIdx.y < tiles0 -> descriptor 0, else descriptor 1 (the gi/gh fusion).
__global__ void gemm_dual_kernel(
    const bf16* __restrict__ A0, int lda0, const bf16* __restrict__ W0, int K0,
    float* __restrict__ C0, int ldc0, int tiles0,
    const bf16* __restrict__ A1, int lda1, const bf16* __restrict__ W1, int K1,
    float* __restrict__ C1, int ldc1) {

    const bf16* A; const bf16* W; float* C;
    int lda, K, ldc, nb;
    if ((int)blockIdx.y < tiles0) {
        A = A0; lda = lda0; W = W0; K = K0; C = C0; ldc = ldc0; nb = blockIdx.y;
    } else {
        A = A1; lda = lda1; W = W1; K = K1; C = C1; ldc = ldc1; nb = blockIdx.y - tiles0;
    }

    const int lane = threadIdx.x & 31;
    const int wave = threadIdx.x >> 5;
    const int msub = wave & 1;        // 0..1  -> 16-row sub-tile
    const int nsub = wave >> 1;       // 0..3  -> 64-col strip
    const int nlo  = lane & 15;
    const int hi   = lane >> 4;       // K-half selector per ISA 16-bit layout

    const int m0   = blockIdx.x * 32 + msub * 16;
    const int colB = nb * 256 + nsub * 64;

    // A fragment row for this lane (A: one row striped across VGPRs in a lane)
    const bf16* Arow = A + (long)(m0 + nlo) * lda;

    // B columns (W row-major (N x K) => column n of B is row n of W, contiguous in K)
    const bf16* Wr0 = W + (long)(colB + 0  + nlo) * K;
    const bf16* Wr1 = W + (long)(colB + 16 + nlo) * K;
    const bf16* Wr2 = W + (long)(colB + 32 + nlo) * K;
    const bf16* Wr3 = W + (long)(colB + 48 + nlo) * K;

    v8f acc0 = {}, acc1 = {}, acc2 = {}, acc3 = {};

    for (int k0 = 0; k0 < K; k0 += 32) {
        v8bf alo = *reinterpret_cast<const v8bf*>(Arow + k0 + hi * 8);
        v8bf ahi = *reinterpret_cast<const v8bf*>(Arow + k0 + 16 + hi * 8);
        v16bf a = __builtin_shufflevector(alo, ahi, 0,1,2,3,4,5,6,7,8,9,10,11,12,13,14,15);

        v8bf b0l = *reinterpret_cast<const v8bf*>(Wr0 + k0 + hi * 8);
        v8bf b0h = *reinterpret_cast<const v8bf*>(Wr0 + k0 + 16 + hi * 8);
        v8bf b1l = *reinterpret_cast<const v8bf*>(Wr1 + k0 + hi * 8);
        v8bf b1h = *reinterpret_cast<const v8bf*>(Wr1 + k0 + 16 + hi * 8);
        v8bf b2l = *reinterpret_cast<const v8bf*>(Wr2 + k0 + hi * 8);
        v8bf b2h = *reinterpret_cast<const v8bf*>(Wr2 + k0 + 16 + hi * 8);
        v8bf b3l = *reinterpret_cast<const v8bf*>(Wr3 + k0 + hi * 8);
        v8bf b3h = *reinterpret_cast<const v8bf*>(Wr3 + k0 + 16 + hi * 8);

        v16bf b0 = __builtin_shufflevector(b0l, b0h, 0,1,2,3,4,5,6,7,8,9,10,11,12,13,14,15);
        v16bf b1 = __builtin_shufflevector(b1l, b1h, 0,1,2,3,4,5,6,7,8,9,10,11,12,13,14,15);
        v16bf b2 = __builtin_shufflevector(b2l, b2h, 0,1,2,3,4,5,6,7,8,9,10,11,12,13,14,15);
        v16bf b3 = __builtin_shufflevector(b3l, b3h, 0,1,2,3,4,5,6,7,8,9,10,11,12,13,14,15);

        acc0 = __builtin_amdgcn_wmma_f32_16x16x32_bf16(false, a, false, b0, (short)0, acc0, false, false);
        acc1 = __builtin_amdgcn_wmma_f32_16x16x32_bf16(false, a, false, b1, (short)0, acc1, false, false);
        acc2 = __builtin_amdgcn_wmma_f32_16x16x32_bf16(false, a, false, b2, (short)0, acc2, false, false);
        acc3 = __builtin_amdgcn_wmma_f32_16x16x32_bf16(false, a, false, b3, (short)0, acc3, false, false);
    }

    // C/D layout: VGPR v -> row m0 + 8*hi + v, column = colB + j*16 + nlo
    #pragma unroll
    for (int v = 0; v < 8; ++v) {
        long row = m0 + hi * 8 + v;
        float* cr = C + row * (long)ldc;
        cr[colB + 0  + nlo] = acc0[v];
        cr[colB + 16 + nlo] = acc1[v];
        cr[colB + 32 + nlo] = acc2[v];
        cr[colB + 48 + nlo] = acc3[v];
    }
}

// ---------------- GRU gate pointwise ----------------
__global__ void gru_gate_kernel(const float* __restrict__ gi, const float* __restrict__ gh,
                                const float* __restrict__ bih, const float* __restrict__ bhh,
                                float* __restrict__ h, unsigned short* __restrict__ hbf) {
    int idx = blockIdx.x * blockDim.x + threadIdx.x;
    if (idx >= N_BATCH * HIDN) return;
    int b = idx >> 8;
    int j = idx & 255;
    long ro = (long)b * G3;
    float ir = gi[ro + j]            + bih[j];
    float iz = gi[ro + HIDN + j]     + bih[HIDN + j];
    float in = gi[ro + 2 * HIDN + j] + bih[2 * HIDN + j];
    float hr = gh[ro + j]            + bhh[j];
    float hz = gh[ro + HIDN + j]     + bhh[HIDN + j];
    float hn = gh[ro + 2 * HIDN + j] + bhh[2 * HIDN + j];
    float r = 1.0f / (1.0f + __expf(-(ir + hr)));
    float z = 1.0f / (1.0f + __expf(-(iz + hz)));
    float n = tanhf(in + r * hn);
    float hnew = (1.0f - z) * n + z * h[idx];
    h[idx]   = hnew;
    hbf[idx] = f32_to_bf16_rne(hnew);
}

// ---------------- head2: smelu-on-the-fly + tiny dense -> output ----------------
__global__ void head2_kernel(const float* __restrict__ act,     // (N, 256): [th(128) | cl(128)]
                             const float* __restrict__ th_b0, const float* __restrict__ cl_b0,
                             const float* __restrict__ th_W1, const float* __restrict__ th_b1,
                             const float* __restrict__ cl_W1, const float* __restrict__ cl_b1,
                             float* __restrict__ out, int t) {
    int idx = blockIdx.x * blockDim.x + threadIdx.x;
    if (idx >= N_BATCH * FOUTN) return;
    int b = idx / FOUTN;
    int o = idx - b * FOUTN;
    const float* ar = act + (long)b * 256;
    float s;
    if (o < 2) {
        s = th_b1[o];
        const float* w = th_W1 + o * REG_H;
        #pragma unroll 4
        for (int k = 0; k < REG_H; ++k)
            s += smelu_f(ar[k] + th_b0[k]) * w[k];
    } else {
        int oo = o - 2;
        s = cl_b1[oo];
        const float* w = cl_W1 + oo * REG_H;
        #pragma unroll 4
        for (int k = 0; k < REG_H; ++k)
            s += smelu_f(ar[REG_H + k] + cl_b0[k]) * w[k];
    }
    out[((long)b * LY + t) * FOUTN + o] = s;
}

// ---------------- launcher ----------------
extern "C" void kernel_launch(void* const* d_in, const int* in_sizes, int n_in,
                              void* d_out, int out_size, void* d_ws, size_t ws_size,
                              hipStream_t stream) {
    (void)in_sizes; (void)n_in; (void)out_size; (void)ws_size;
    const float* x_f     = (const float*)d_in[0];
    const float* y_t     = (const float*)d_in[2];
    const float* p       = (const float*)d_in[3];
    const float* enc_Wih = (const float*)d_in[4];
    const float* enc_Whh = (const float*)d_in[5];
    const float* enc_bih = (const float*)d_in[6];
    const float* enc_bhh = (const float*)d_in[7];
    const float* dec_Wih = (const float*)d_in[8];
    const float* dec_Whh = (const float*)d_in[9];
    const float* dec_bih = (const float*)d_in[10];
    const float* dec_bhh = (const float*)d_in[11];
    const float* th_W0   = (const float*)d_in[12];
    const float* th_b0   = (const float*)d_in[13];
    const float* th_W1   = (const float*)d_in[14];
    const float* th_b1   = (const float*)d_in[15];
    const float* cl_W0   = (const float*)d_in[16];
    const float* cl_b0   = (const float*)d_in[17];
    const float* cl_W1   = (const float*)d_in[18];
    const float* cl_b1   = (const float*)d_in[19];
    float* out = (float*)d_out;

    // ---- workspace carve-out (all 256B aligned) ----
    char* ws = (char*)d_ws;
    size_t off = 0;
    auto alloc = [&](size_t bytes) -> void* {
        void* r = ws + off;
        off = (off + bytes + 255) & ~(size_t)255;
        return r;
    };
    float*          h_f32   = (float*)alloc((size_t)N_BATCH * HIDN * 4);
    unsigned short* h_bf    = (unsigned short*)alloc((size_t)N_BATCH * HIDN * 2);
    float*          gg_h    = (float*)alloc((size_t)N_BATCH * G3 * 4);
    float*          gg_i    = (float*)alloc((size_t)N_BATCH * G3 * 4);
    float*          head1   = (float*)alloc((size_t)N_BATCH * 256 * 4);
    unsigned short* xf_bf   = (unsigned short*)alloc((size_t)N_BATCH * LX * FF * 2);
    unsigned short* dec_bf  = (unsigned short*)alloc((size_t)LY * N_BATCH * DEC_INW * 2);
    unsigned short* eWih_bf = (unsigned short*)alloc((size_t)G3 * FF * 2);
    unsigned short* eWhh_bf = (unsigned short*)alloc((size_t)G3 * HIDN * 2);
    unsigned short* dWih_bf = (unsigned short*)alloc((size_t)G3 * DEC_INW * 2);
    unsigned short* dWhh_bf = (unsigned short*)alloc((size_t)G3 * HIDN * 2);
    unsigned short* w0_bf   = (unsigned short*)alloc((size_t)256 * HIDN * 2);  // [th_W0 ; cl_W0]

    auto cvt = [&](const float* s, unsigned short* d, long n) {
        int g = (int)((n + 255) / 256);
        cvt_bf16_kernel<<<g, 256, 0, stream>>>(s, d, n);
    };

    // one-time prep
    cvt(x_f,     xf_bf,   (long)N_BATCH * LX * FF);
    cvt(enc_Wih, eWih_bf, (long)G3 * FF);
    cvt(enc_Whh, eWhh_bf, (long)G3 * HIDN);
    cvt(dec_Wih, dWih_bf, (long)G3 * DEC_INW);
    cvt(dec_Whh, dWhh_bf, (long)G3 * HIDN);
    cvt(th_W0,   w0_bf,                (long)REG_H * HIDN);
    cvt(cl_W0,   w0_bf + (size_t)REG_H * HIDN, (long)REG_H * HIDN);
    {
        long n = (long)LY * N_BATCH * DEC_INW;
        pack_dec_kernel<<<(int)((n + 255) / 256), 256, 0, stream>>>(y_t, p, dec_bf);
    }
    zero_h_kernel<<<(N_BATCH * HIDN) / 256, 256, 0, stream>>>(h_f32, h_bf);

    const dim3 gemmBlk(256);
    const dim3 gruGrid(N_BATCH / 32, 6);   // 3 tiles for gh + 3 tiles for gi
    const int  gateGrid = (N_BATCH * HIDN) / 256;

    // ---- encoder: 168 sequential GRU steps ----
    for (int t = 0; t < LX; ++t) {
        gemm_dual_kernel<<<gruGrid, gemmBlk, 0, stream>>>(
            (const bf16*)h_bf, HIDN, (const bf16*)eWhh_bf, HIDN, gg_h, G3, 3,
            (const bf16*)(xf_bf + (size_t)t * FF), LX * FF, (const bf16*)eWih_bf, FF, gg_i, G3);
        gru_gate_kernel<<<gateGrid, 256, 0, stream>>>(gg_i, gg_h, enc_bih, enc_bhh, h_f32, h_bf);
    }

    // ---- decoder: 72 sequential steps with fused heads ----
    const dim3 headGrid(N_BATCH / 32, 1);
    const int  h2Grid = (N_BATCH * FOUTN + 255) / 256;
    for (int t = 0; t < LY; ++t) {
        gemm_dual_kernel<<<gruGrid, gemmBlk, 0, stream>>>(
            (const bf16*)h_bf, HIDN, (const bf16*)dWhh_bf, HIDN, gg_h, G3, 3,
            (const bf16*)(dec_bf + (size_t)t * N_BATCH * DEC_INW), DEC_INW,
            (const bf16*)dWih_bf, DEC_INW, gg_i, G3);
        gru_gate_kernel<<<gateGrid, 256, 0, stream>>>(gg_i, gg_h, dec_bih, dec_bhh, h_f32, h_bf);
        // fused [th_W0 ; cl_W0] projection: (2048x256) @ (256x256)^T
        gemm_dual_kernel<<<headGrid, gemmBlk, 0, stream>>>(
            (const bf16*)h_bf, HIDN, (const bf16*)w0_bf, HIDN, head1, 256, 1,
            (const bf16*)h_bf, HIDN, (const bf16*)w0_bf, HIDN, head1, 256);
        head2_kernel<<<h2Grid, 256, 0, stream>>>(head1, th_b0, cl_b0,
                                                 th_W1, th_b1, cl_W1, cl_b1, out, t);
    }
}
